// RWKV_TimeMix_37864431682654
// MI455X (gfx1250) — matrix operand order repaired and checked by reading, not compiled, not run on hardware
//
#include <hip/hip_runtime.h>
#include <hip/hip_bf16.h>

typedef __bf16 bf16;
typedef __attribute__((ext_vector_type(16))) __bf16 v16bf;
typedef __attribute__((ext_vector_type(8)))  float  v8f;

union Frag {
    v16bf v;
    bf16  e[16];
    uint4 q[2];
};

static __device__ __forceinline__ bf16 f2b(float x) { return (bf16)x; }
static __device__ __forceinline__ float b2f(bf16 x) { return (float)x; }

// ---------------------------------------------------------------------------
// CDNA5 async global->LDS copy (16B per lane), tracked by ASYNCcnt.
// GV addressing: LDS address taken from the vdst VGPR, global address from the 64-bit vaddr.
// ---------------------------------------------------------------------------
static __device__ __forceinline__ void async_copy16(void* lds, const void* gptr) {
    unsigned int  l = (unsigned int)(unsigned long long)(uintptr_t)lds; // low 32b = LDS offset
    unsigned long long g = (unsigned long long)(uintptr_t)gptr;
    asm volatile("global_load_async_to_lds_b128 %0, %1, off"
                 :: "v"(l), "v"(g) : "memory");
}

static __device__ __forceinline__ void wait_async0() {
#if __has_builtin(__builtin_amdgcn_s_wait_asynccnt)
    __builtin_amdgcn_s_wait_asynccnt(0);
#else
    asm volatile("s_wait_asynccnt 0" ::: "memory");
#endif
}

// ---------------------------------------------------------------------------
// Kernel 1: time-shift + bf16 convert.  xs[b,t,c] = (c<512 ? x[b,t-1,c] : x[b,t,c])
// ---------------------------------------------------------------------------
__global__ void xs_kernel(const float* __restrict__ x, bf16* __restrict__ xsb) {
    int i = blockIdx.x * 256 + threadIdx.x;          // i < 8*1024*1024
    int c = i & 1023;
    int t = (i >> 10) & 1023;
    float v;
    if (c < 512) v = (t > 0) ? x[i - 1024] : 0.0f;
    else         v = x[i];
    xsb[i] = f2b(v);
}

// ---------------------------------------------------------------------------
// Kernel 2: transpose+convert weights to (N,K) bf16.
// ---------------------------------------------------------------------------
__global__ void wprep_kernel(const float* __restrict__ Wk, const float* __restrict__ Wv,
                             const float* __restrict__ Wr, const float* __restrict__ Wo,
                             bf16* __restrict__ w3t, bf16* __restrict__ wot) {
    int i = blockIdx.x * 256 + threadIdx.x;          // i < 4*1024*1024
    int n = i >> 10, k = i & 1023;
    if (n < 3072) {
        int seg = n >> 10, nn = n & 1023;
        const float* W = (seg == 0) ? Wk : ((seg == 1) ? Wv : Wr);
        w3t[(size_t)n * 1024 + k] = f2b(W[(size_t)k * 1024 + nn]);
    } else {
        int nn = n - 3072;
        wot[(size_t)nn * 1024 + k] = f2b(Wo[(size_t)k * 1024 + nn]);
    }
}

// ---------------------------------------------------------------------------
// Kernel 3/7: bf16 WMMA GEMM,  C(M,N) = A(M,K) @ Bw^T  with Bw stored (N,K).
// 128 threads (4 waves), tile BM=64 BN=64 BK=32, double-buffered LDS filled
// by global_load_async_to_lds_b128; one workgroup barrier per K-step.
// mode 0: epilogue K=exp(clip(.+bk)), V=.+bv, R=sigmoid(.+br) -> bf16.
// mode 1: epilogue out = (. + bo[n]) * gamma[t] -> f32.
// ---------------------------------------------------------------------------
__global__ void gemm_bf16_kernel(const bf16* __restrict__ A, const bf16* __restrict__ Bw,
                                 int K, int mode,
                                 const float* __restrict__ bias_k, const float* __restrict__ bias_v,
                                 const float* __restrict__ bias_r,
                                 const float* __restrict__ bo, const float* __restrict__ gamma,
                                 bf16* __restrict__ kb, bf16* __restrict__ vb, bf16* __restrict__ rsb,
                                 float* __restrict__ outf) {
    __shared__ bf16 At[2][64 * 32];
    __shared__ bf16 Bt[2][64 * 32];
    const int tid = threadIdx.x;
    const int lane = tid & 31, wave = tid >> 5;
    const int m0 = blockIdx.y * 64, n0 = blockIdx.x * 64;

    v8f acc[4] = {};
    const int nk = K >> 5;
    const int r_ld = tid >> 2, c8_ld = (tid & 3) << 3;   // this thread's 16B slots

    // Prologue: async-stage tile 0 into buffer 0.
#pragma unroll
    for (int i = 0; i < 2; ++i) {
        int r = r_ld + i * 32;
        async_copy16(&At[0][r * 32 + c8_ld], &A[(size_t)(m0 + r) * K + c8_ld]);
        async_copy16(&Bt[0][r * 32 + c8_ld], &Bw[(size_t)(n0 + r) * K + c8_ld]);
    }

    for (int kk = 0; kk < nk; ++kk) {
        const int cur = kk & 1;
        wait_async0();               // this wave's async copies for tile kk done
        __syncthreads();             // => every wave's copies done; prev buffer free

        if (kk + 1 < nk) {           // async-stage tile kk+1 into the other buffer
            const int k1 = (kk + 1) << 5, nxt = cur ^ 1;
#pragma unroll
            for (int i = 0; i < 2; ++i) {
                int r = r_ld + i * 32;
                async_copy16(&At[nxt][r * 32 + c8_ld], &A[(size_t)(m0 + r) * K + k1 + c8_ld]);
                async_copy16(&Bt[nxt][r * 32 + c8_ld], &Bw[(size_t)(n0 + r) * K + k1 + c8_ld]);
            }
        }

        // A fragment: lane row = wave*16 + lane%16; K = (e>=8?16:0)+(lane&16?8:0)+(e&7).
        Frag fa;
        {
            int m = wave * 16 + (lane & 15);
            int c0 = (lane & 16) ? 8 : 0;
            fa.q[0] = *reinterpret_cast<const uint4*>(&At[cur][m * 32 + c0]);
            fa.q[1] = *reinterpret_cast<const uint4*>(&At[cur][m * 32 + 16 + c0]);
        }
#pragma unroll
        for (int ct = 0; ct < 4; ++ct) {
            Frag fb;                 // B layout: K = (e>=8?16:0)+(lane&16?8:0)+(e&7)
            int n = ct * 16 + (lane & 15);
            int us = (lane & 16) ? 8 : 0;
            fb.q[0] = *reinterpret_cast<const uint4*>(&Bt[cur][n * 32 + us]);
            fb.q[1] = *reinterpret_cast<const uint4*>(&Bt[cur][n * 32 + us + 16]);
            acc[ct] = __builtin_amdgcn_wmma_f32_16x16x32_bf16(
                false, fa.v, false, fb.v, (short)0, acc[ct], false, false);
        }
    }

    // Epilogue (C/D layout: VGPR r -> M = r + (lane>=16 ? 8 : 0), N = lane%16).
#pragma unroll
    for (int ct = 0; ct < 4; ++ct) {
        int n = n0 + ct * 16 + (lane & 15);
#pragma unroll
        for (int r = 0; r < 8; ++r) {
            int m = m0 + wave * 16 + r + ((lane & 16) ? 8 : 0);
            float val = acc[ct][r];
            if (mode == 0) {
                int seg = n >> 10, nn = n & 1023;
                size_t o = (size_t)m * 1024 + nn;
                if (seg == 0) {
                    float kx = val + bias_k[nn];
                    kx = fminf(fmaxf(kx, -60.0f), 30.0f);
                    kb[o] = f2b(__expf(kx));
                } else if (seg == 1) {
                    vb[o] = f2b(val + bias_v[nn]);
                } else {
                    float rx = val + bias_r[nn];
                    rsb[o] = f2b(1.0f / (1.0f + __expf(-rx)));
                }
            } else {
                int t = m & 1023;
                outf[(size_t)m * 1024 + n] = (val + bo[n]) * gamma[t];
            }
        }
    }
}

// ---------------------------------------------------------------------------
// Kernel 4: cumsum of exp-k over t (f32 accumulate), coalesced across channels.
// ---------------------------------------------------------------------------
__global__ void cumsum_kernel(const bf16* __restrict__ kbuf, float* __restrict__ sumk) {
    int c = blockIdx.x * 256 + threadIdx.x;          // channel 0..1023
    int b = blockIdx.y;
    float acc = 0.0f;
    size_t base = (size_t)b * 1024 * 1024 + c;
    for (int t = 0; t < 1024; ++t) {
        acc += b2f(kbuf[base + (size_t)t * 1024]);
        sumk[base + (size_t)t * 1024] = acc;
    }
}

// ---------------------------------------------------------------------------
// Kernel 5: wkv[b,t,h,c] = beta[h,t] * sum_{u<=t} time_w[h,1023-(t-u)]*alpha[h,u]*k[u]*v[u]
// ---------------------------------------------------------------------------
__global__ void wkv_kernel(const bf16* __restrict__ kbuf, const bf16* __restrict__ vbuf,
                           const float* __restrict__ time_w, const float* __restrict__ alpha,
                           const float* __restrict__ beta, bf16* __restrict__ wkvb) {
    __shared__ float twl[1024];
    __shared__ float all[1024];
    __shared__ bf16  kvT[64 * 32];                   // (c=64) x (u=32), row-major

    const int tid = threadIdx.x, lane = tid & 31, wave = tid >> 5;
    const int gx = blockIdx.x, h = blockIdx.y, b = blockIdx.z;

    // Verbatim 4KB row copies -> async to LDS (2 x 16B chunks per thread per row).
#pragma unroll
    for (int i = 0; i < 2; ++i) {
        int f0 = (tid + i * 128) * 4;                // float index, 4 floats = 16B
        async_copy16(&twl[f0], &time_w[h * 1024 + f0]);
        async_copy16(&all[f0], &alpha[h * 1024 + f0]);
    }
    wait_async0();
    __syncthreads();

    const int t0b = gx * 64;
    const int t0w = t0b + wave * 16;
    v8f acc[4] = {};
    const int nchunk = (t0b >> 5) + 2;               // u chunks of 32 up to t0b+63

    for (int cc = 0; cc < nchunk; ++cc) {
        const int u0 = cc << 5;
        // Stage kv chunk transposed: each thread handles 2 (u, c-octet) groups.
#pragma unroll
        for (int ii = 0; ii < 2; ++ii) {
            int idx = tid + ii * 128;                // 256 groups
            int u = idx >> 3, c0 = (idx & 7) << 3;
            size_t g = (size_t)(b * 1024 + u0 + u) * 1024 + h * 64 + c0;
            uint4 kq = *reinterpret_cast<const uint4*>(&kbuf[g]);
            uint4 vq = *reinterpret_cast<const uint4*>(&vbuf[g]);
            const bf16* kp = reinterpret_cast<const bf16*>(&kq);
            const bf16* vp = reinterpret_cast<const bf16*>(&vq);
#pragma unroll
            for (int j = 0; j < 8; ++j)
                kvT[(c0 + j) * 32 + u] = f2b(b2f(kp[j]) * b2f(vp[j]));
        }
        __syncthreads();

        if (u0 <= t0w + 15) {                        // causal: chunk touches this wave's rows
            Frag fa;
            const int tm = t0w + (lane & 15);
#pragma unroll
            for (int e = 0; e < 16; ++e) {           // A layout K-mapping
                int u = u0 + ((e >= 8) ? 16 : 0) + ((lane & 16) ? 8 : 0) + (e & 7);
                int d = tm - u;
                float wv = (d >= 0) ? twl[1023 - d] * all[u] : 0.0f;
                fa.e[e] = f2b(wv);
            }
#pragma unroll
            for (int ct = 0; ct < 4; ++ct) {
                Frag fb;
                int c = ct * 16 + (lane & 15);
                int us = (lane & 16) ? 8 : 0;
                fb.q[0] = *reinterpret_cast<const uint4*>(&kvT[c * 32 + us]);
                fb.q[1] = *reinterpret_cast<const uint4*>(&kvT[c * 32 + us + 16]);
                acc[ct] = __builtin_amdgcn_wmma_f32_16x16x32_bf16(
                    false, fa.v, false, fb.v, (short)0, acc[ct], false, false);
            }
        }
        __syncthreads();
    }

#pragma unroll
    for (int ct = 0; ct < 4; ++ct) {
        int c = ct * 16 + (lane & 15);
#pragma unroll
        for (int r = 0; r < 8; ++r) {
            int t = t0w + r + ((lane & 16) ? 8 : 0);
            float val = acc[ct][r] * beta[h * 1024 + t];
            wkvb[(size_t)(b * 1024 + t) * 1024 + h * 64 + c] = f2b(val);
        }
    }
}

// ---------------------------------------------------------------------------
// Kernel 6: rwkv = sigmoid(r) * wkv / sum_k  (bf16 out for final GEMM)
// ---------------------------------------------------------------------------
__global__ void rwkv_kernel(const bf16* __restrict__ rsb, const bf16* __restrict__ wkvb,
                            const float* __restrict__ sumk, bf16* __restrict__ rwkvb) {
    int i = blockIdx.x * 256 + threadIdx.x;          // i < 8*1024*1024
    rwkvb[i] = f2b(b2f(rsb[i]) * b2f(wkvb[i]) / sumk[i]);
}

// ---------------------------------------------------------------------------
extern "C" void kernel_launch(void* const* d_in, const int* in_sizes, int n_in,
                              void* d_out, int out_size, void* d_ws, size_t ws_size,
                              hipStream_t stream) {
    (void)in_sizes; (void)n_in; (void)out_size; (void)ws_size;
    const float* x     = (const float*)d_in[0];
    const float* tw    = (const float*)d_in[1];
    const float* alpha = (const float*)d_in[2];
    const float* beta  = (const float*)d_in[3];
    const float* gamma = (const float*)d_in[4];
    const float* Wk    = (const float*)d_in[5];
    const float* bk    = (const float*)d_in[6];
    const float* Wv    = (const float*)d_in[7];
    const float* bv    = (const float*)d_in[8];
    const float* Wr    = (const float*)d_in[9];
    const float* br    = (const float*)d_in[10];
    const float* Wo    = (const float*)d_in[11];
    const float* bo    = (const float*)d_in[12];
    float* out = (float*)d_out;

    const size_t M = 8192;            // B*T
    char* ws = (char*)d_ws;
    size_t off = 0;
    bf16* xsb   = (bf16*)(ws + off); off += M * 1024 * 2;            // 16 MB
    bf16* w3t   = (bf16*)(ws + off); off += (size_t)3072 * 1024 * 2; // 6 MB
    bf16* wot   = (bf16*)(ws + off); off += (size_t)1024 * 1024 * 2; // 2 MB
    bf16* kb    = (bf16*)(ws + off); off += M * 1024 * 2;            // 16 MB
    bf16* vb    = (bf16*)(ws + off); off += M * 1024 * 2;            // 16 MB
    bf16* rsb   = (bf16*)(ws + off); off += M * 1024 * 2;            // 16 MB
    bf16* wkvb  = (bf16*)(ws + off); off += M * 1024 * 2;            // 16 MB
    bf16* rwkvb = (bf16*)(ws + off); off += M * 1024 * 2;            // 16 MB
    float* sumk = (float*)(ws + off); off += M * 1024 * 4;           // 32 MB

    xs_kernel<<<(M * 1024) / 256, 256, 0, stream>>>(x, xsb);
    wprep_kernel<<<(4 * 1024 * 1024) / 256, 256, 0, stream>>>(Wk, Wv, Wr, Wo, w3t, wot);
    gemm_bf16_kernel<<<dim3(3072 / 64, M / 64), 128, 0, stream>>>(
        xsb, w3t, 1024, 0, bk, bv, br, nullptr, nullptr, kb, vb, rsb, nullptr);
    cumsum_kernel<<<dim3(4, 8), 256, 0, stream>>>(kb, sumk);
    wkv_kernel<<<dim3(16, 16, 8), 128, 0, stream>>>(kb, vb, tw, alpha, beta, wkvb);
    rwkv_kernel<<<(M * 1024) / 256, 256, 0, stream>>>(rsb, wkvb, sumk, rwkvb);
    gemm_bf16_kernel<<<dim3(1024 / 64, M / 64), 128, 0, stream>>>(
        rwkvb, wot, 1024, 1, nullptr, nullptr, nullptr, bo, gamma,
        nullptr, nullptr, nullptr, out);
}
